// GQAttention_88192858456722
// MI455X (gfx1250) — compile-verified, matrix operand order checked by reading, not run on hardware
//
#include <hip/hip_runtime.h>
#include <math.h>

typedef __bf16 bf16;
typedef __attribute__((ext_vector_type(16))) __bf16 v16bf;
typedef __attribute__((ext_vector_type(8)))  __bf16 v8bf;
typedef __attribute__((ext_vector_type(8)))  float  v8f;

constexpr int Bn = 2, Sn = 2048, HIDn = 2048, Hn = 32, KVn = 4, Dn = 64;
constexpr int GROUPn = Hn / KVn;   // 8 query heads per kv head

// ---------------------------------------------------------------------------
// WMMA helpers
// ---------------------------------------------------------------------------
__device__ __forceinline__ v8f wmma_bf16(v16bf a, v16bf b, v8f c) {
  // 8 args: (neg_a, A, neg_b, B, c_mod, C, reuse_a, reuse_b)
  return __builtin_amdgcn_wmma_f32_16x16x32_bf16(false, a, false, b, (short)0, c,
                                                 false, false);
}

// Load a 16x32 bf16 A/B fragment from row-major memory (row stride `stride`
// elements).  Per the CDNA5 ISA 16-bit A-matrix layout: lanes 0-15 hold
// K={0..7,16..23}, lanes 16-31 hold K={8..15,24..31} of row (lane&15).
__device__ __forceinline__ v16bf load_frag_g(const bf16* base, int stride, int lane) {
  const int row = lane & 15;
  const int hi  = lane >> 4;
  const bf16* p = base + row * stride + hi * 8;
  v8bf lo = *(const v8bf*)p;          // K = hi*8 .. hi*8+7
  v8bf hp = *(const v8bf*)(p + 16);   // K = 16 + hi*8 .. 16 + hi*8+7
  v16bf r;
#pragma unroll
  for (int i = 0; i < 8; ++i) { r[i] = lo[i]; r[i + 8] = hp[i]; }
  return r;
}

// Fully unrolled, depth-2 software-pipelined 16x16 GEMM tile over K=HIDn.
// Full unroll removes all loop-carried fragment phis, so every load coalesces
// directly into its WMMA operand registers (no copy movs), and since
// HIDn*2 bytes fits the 24-bit instruction offset, every load is an
// immediate-offset access off a single base address (no per-chunk VALU math).
__device__ __forceinline__ v8f gemm_tile(const bf16* __restrict__ ab,
                                         const bf16* __restrict__ bb, int lane) {
  constexpr int NC = HIDn / 32;      // 64 K-chunks
  v8f acc = {0.f, 0.f, 0.f, 0.f, 0.f, 0.f, 0.f, 0.f};
  __builtin_prefetch(ab + 16 * HIDn, 0, 3);  // speculative, WGP-side
  __builtin_prefetch(bb + 16 * HIDn, 0, 3);
  v16bf af[2], bg[2];
  af[0] = load_frag_g(ab,      HIDn, lane);
  bg[0] = load_frag_g(bb,      HIDn, lane);
  af[1] = load_frag_g(ab + 32, HIDn, lane);
  bg[1] = load_frag_g(bb + 32, HIDn, lane);
#pragma unroll
  for (int i = 0; i < NC; ++i) {
    const int cur = i & 1;
    acc = wmma_bf16(af[cur], bg[cur], acc);
    if (i + 2 < NC) {                 // compile-time resolved under unroll
      af[cur] = load_frag_g(ab + (i + 2) * 32, HIDn, lane);
      bg[cur] = load_frag_g(bb + (i + 2) * 32, HIDn, lane);
    }
  }
  return acc;
}

// ---------------------------------------------------------------------------
// Stage 0: dtype conversion / weight transpose
// ---------------------------------------------------------------------------
__global__ void __launch_bounds__(256) cvt_bf16_kernel(const float* __restrict__ in,
                                                       bf16* __restrict__ out, int n) {
  int i = blockIdx.x * 256 + threadIdx.x;
  if (i < n) out[i] = (bf16)in[i];
}

// W is [K,N] fp32 row-major -> WT is [N,K] bf16 row-major
__global__ void __launch_bounds__(256) transpose_bf16_kernel(const float* __restrict__ W,
                                                             bf16* __restrict__ WT,
                                                             int K, int N) {
  int i = blockIdx.x * 256 + threadIdx.x;
  if (i < K * N) {
    int nn = i / K;
    int kk = i - nn * K;
    WT[i] = (bf16)W[kk * N + nn];
  }
}

// ---------------------------------------------------------------------------
// Stage 1: fused QKV projection + RoPE.
// One wave computes one 16x16 output tile of q (128 tiles), k (16) or v (16).
// ---------------------------------------------------------------------------
__global__ void __launch_bounds__(256) qkv_rope_kernel(
    const bf16* __restrict__ xbf,  const bf16* __restrict__ wqT,
    const bf16* __restrict__ wkT,  const bf16* __restrict__ wvT,
    const float* __restrict__ cosT, const float* __restrict__ sinT,
    bf16* __restrict__ Qbf, bf16* __restrict__ Kbf, bf16* __restrict__ Vt) {
  const int lane = threadIdx.x & 31;
  const int wid  = threadIdx.x >> 5;
  const int gid  = blockIdx.x * 8 + wid;
  const int mt   = gid / 160;           // 256 row tiles over B*S
  const int nt   = gid % 160;           // 128 q + 16 k + 16 v column tiles
  const int m0   = mt * 16;
  const int b    = m0 / Sn;
  const int s0   = m0 - b * Sn;
  const int col  = lane & 15;
  const int hi   = lane >> 4;

  const bf16* wT;
  int n0;
  if (nt < 128)      { wT = wqT; n0 = nt * 16; }
  else if (nt < 144) { wT = wkT; n0 = (nt - 128) * 16; }
  else               { wT = wvT; n0 = (nt - 144) * 16; }

  v8f acc = gemm_tile(xbf + (size_t)m0 * HIDn, wT + (size_t)n0 * HIDn, lane);

  if (nt < 144) {
    // q or k tile: apply RoPE then store [b][head][s][d] bf16.
    bf16* outp;
    int d0;
    if (nt < 128) { int h = nt >> 2;            d0 = (nt & 3) * 16;
                    outp = Qbf + ((size_t)(b * Hn  + h ) * Sn) * Dn; }
    else          { int n2 = nt - 128; int kv = n2 >> 2; d0 = (n2 & 3) * 16;
                    outp = Kbf + ((size_t)(b * KVn + kv) * Sn) * Dn; }
    const int dcol = d0 + col;
    const int ii   = dcol >> 1;
    const bool odd = (dcol & 1) != 0;
#pragma unroll
    for (int r = 0; r < 8; ++r) {
      int srow   = s0 + hi * 8 + r;
      float c    = acc[r];
      float prt  = __shfl_xor(c, 1, 32);       // RoPE partner (adjacent d)
      float cv   = cosT[srow * (Dn / 2) + ii];
      float sv   = sinT[srow * (Dn / 2) + ii];
      float res  = odd ? (prt * sv + c * cv) : (c * cv - prt * sv);
      outp[(size_t)srow * Dn + dcol] = (bf16)res;
    }
  } else {
    // v tile: store transposed [b][kv][d][s] so PV B-fragments are contiguous.
    int n2 = nt - 144;
    int kv = n2 >> 2;
    int d  = (n2 & 3) * 16 + col;
    bf16* vp = Vt + ((size_t)(b * KVn + kv) * Dn + d) * Sn + s0 + hi * 8;
    v8bf pk;
#pragma unroll
    for (int r = 0; r < 8; ++r) pk[r] = (bf16)acc[r];
    *(v8bf*)vp = pk;   // one b128 store of 8 consecutive seq positions
  }
}

// ---------------------------------------------------------------------------
// Stage 2: causal flash attention.  One wave per (b, h, 16-query tile);
// streams keys in blocks of 32 with online softmax.  All K and V fragment
// loads for a tile are issued together at the top of the body: the QK^T
// WMMAs need only the K loads (partial loadcnt wait), the V loads drain under
// the softmax, and next-tile loads overlap the previous PV WMMAs.  No
// loop-carried fragments -> no register-copy movs.
// ---------------------------------------------------------------------------
__global__ void __launch_bounds__(128) flash_attn_kernel(
    const bf16* __restrict__ Qbf, const bf16* __restrict__ Kbf,
    const bf16* __restrict__ Vt,  bf16* __restrict__ Ao) {
  __shared__ __align__(16) bf16 plds[4][16 * 32];

  const int lane = threadIdx.x & 31;
  const int wid  = threadIdx.x >> 5;
  const int gid  = blockIdx.x * 4 + wid;
  const int QT   = Sn / 16;                 // 128
  const int qt   = gid % QT;
  const int h    = (gid / QT) % Hn;
  const int b    = gid / (QT * Hn);
  const int kv   = h / GROUPn;
  const int q0   = qt * 16;
  const int col  = lane & 15;
  const int hi   = lane >> 4;

  const bf16* qbase = Qbf + ((size_t)(b * Hn + h) * Sn + q0) * Dn;
  const v16bf qa0 = load_frag_g(qbase,      Dn, lane);   // d = 0..31
  const v16bf qa1 = load_frag_g(qbase + 32, Dn, lane);   // d = 32..63

  const bf16* kroot = Kbf + ((size_t)(b * KVn + kv) * Sn) * Dn;
  const bf16* vroot = Vt  + ((size_t)(b * KVn + kv) * Dn) * Sn;

  float mrow[8], lrow[8];
  v8f o0 = {0,0,0,0,0,0,0,0}, o1 = o0, o2 = o0, o3 = o0;
#pragma unroll
  for (int r = 0; r < 8; ++r) { mrow[r] = -INFINITY; lrow[r] = 0.f; }

  const int tEnd = q0 + 16;
  for (int t0 = 0; t0 < tEnd; t0 += 32) {
    // ---- issue all 16 loads for this tile: K first, then V ----
    const bf16* kb = kroot + (size_t)t0 * Dn;
    v16bf k00 = load_frag_g(kb,                Dn, lane);
    v16bf k01 = load_frag_g(kb + 32,           Dn, lane);
    v16bf k10 = load_frag_g(kb + 16 * Dn,      Dn, lane);
    v16bf k11 = load_frag_g(kb + 16 * Dn + 32, Dn, lane);
    v16bf vb0 = load_frag_g(vroot +  0 * Sn + t0, Sn, lane);
    v16bf vb1 = load_frag_g(vroot + 16 * Sn + t0, Sn, lane);
    v16bf vb2 = load_frag_g(vroot + 32 * Sn + t0, Sn, lane);
    v16bf vb3 = load_frag_g(vroot + 48 * Sn + t0, Sn, lane);

    // ---- scores: S = Q K^T * 1/sqrt(D), two 16x16 tiles over 32 keys ----
    v8f sA = {0,0,0,0,0,0,0,0}, sB = sA;
    sA = wmma_bf16(qa0, k00, sA);
    sA = wmma_bf16(qa1, k01, sA);
    sB = wmma_bf16(qa0, k10, sB);
    sB = wmma_bf16(qa1, k11, sB);

    // ---- online softmax over the 32-key block (V loads drain underneath) --
#pragma unroll
    for (int r = 0; r < 8; ++r) {
      int srow = q0 + hi * 8 + r;
      float e0 = sA[r] * 0.125f;
      float e1 = sB[r] * 0.125f;
      if (t0 + col      > srow) e0 = -INFINITY;   // causal mask
      if (t0 + 16 + col > srow) e1 = -INFINITY;
      float rm = fmaxf(e0, e1);
#pragma unroll
      for (int off = 1; off < 16; off <<= 1) rm = fmaxf(rm, __shfl_xor(rm, off, 16));
      float nm   = fmaxf(mrow[r], rm);            // rm is finite (t0 <= srow)
      float p0   = __expf(e0 - nm);
      float p1   = __expf(e1 - nm);
      float corr = __expf(mrow[r] - nm);
      float ps   = p0 + p1;
#pragma unroll
      for (int off = 1; off < 16; off <<= 1) ps += __shfl_xor(ps, off, 16);
      lrow[r] = lrow[r] * corr + ps;
      mrow[r] = nm;
      o0[r] *= corr; o1[r] *= corr; o2[r] *= corr; o3[r] *= corr;
      int row = hi * 8 + r;
      plds[wid][row * 32 + col]      = (bf16)p0;  // stage P for transpose
      plds[wid][row * 32 + 16 + col] = (bf16)p1;
    }

    // ---- P fragment (A-layout) from LDS, then O += P V ----
    const bf16* pr = &plds[wid][(lane & 15) * 32 + hi * 8];
    v8bf lo = *(const v8bf*)pr;
    v8bf hp = *(const v8bf*)(pr + 16);
    v16bf pa;
#pragma unroll
    for (int i = 0; i < 8; ++i) { pa[i] = lo[i]; pa[i + 8] = hp[i]; }

    o0 = wmma_bf16(pa, vb0, o0);
    o1 = wmma_bf16(pa, vb1, o1);
    o2 = wmma_bf16(pa, vb2, o2);
    o3 = wmma_bf16(pa, vb3, o3);
  }

  // ---- normalize and store attention output [b][s][h*64+d] as bf16 ----
  bf16* aout = Ao + (size_t)b * Sn * (Hn * Dn) + h * Dn;
#pragma unroll
  for (int r = 0; r < 8; ++r) {
    int srow  = q0 + hi * 8 + r;
    float inv = 1.0f / lrow[r];
    bf16* op  = aout + (size_t)srow * (Hn * Dn);
    op[ 0 + col] = (bf16)(o0[r] * inv);
    op[16 + col] = (bf16)(o1[r] * inv);
    op[32 + col] = (bf16)(o2[r] * inv);
    op[48 + col] = (bf16)(o3[r] * inv);
  }
}

// ---------------------------------------------------------------------------
// Stage 3: output projection (B*S,2048) x (2048,2048) -> fp32 d_out
// ---------------------------------------------------------------------------
__global__ void __launch_bounds__(256) outproj_kernel(
    const bf16* __restrict__ Ao, const bf16* __restrict__ woT,
    float* __restrict__ out) {
  const int lane = threadIdx.x & 31;
  const int wid  = threadIdx.x >> 5;
  const int gid  = blockIdx.x * 8 + wid;
  const int mt   = gid >> 7;            // 256 row tiles
  const int nt   = gid & 127;           // 128 col tiles
  const int m0   = mt * 16, n0 = nt * 16;
  const int col  = lane & 15, hi = lane >> 4;

  v8f acc = gemm_tile(Ao + (size_t)m0 * HIDn, woT + (size_t)n0 * HIDn, lane);

#pragma unroll
  for (int r = 0; r < 8; ++r) {
    int row = m0 + hi * 8 + r;
    out[(size_t)row * HIDn + n0 + col] = acc[r];
  }
}

// ---------------------------------------------------------------------------
// Host launcher
// ---------------------------------------------------------------------------
extern "C" void kernel_launch(void* const* d_in, const int* in_sizes, int n_in,
                              void* d_out, int out_size, void* d_ws, size_t ws_size,
                              hipStream_t stream) {
  const float* x    = (const float*)d_in[0];
  const float* cosT = (const float*)d_in[1];
  const float* sinT = (const float*)d_in[2];
  const float* wq   = (const float*)d_in[3];
  const float* wk   = (const float*)d_in[4];
  const float* wv   = (const float*)d_in[5];
  const float* wo   = (const float*)d_in[6];
  float* out = (float*)d_out;

  char* ws = (char*)d_ws;
  size_t off = 0;
  auto alloc = [&](size_t bytes) -> void* {
    void* p = ws + off;
    off += (bytes + 255) & ~(size_t)255;
    return p;
  };
  const size_t NX = (size_t)Bn * Sn * HIDn;          // 8.4M elements
  bf16* xbf = (bf16*)alloc(NX * 2);
  bf16* wqT = (bf16*)alloc((size_t)HIDn * HIDn * 2);
  bf16* wkT = (bf16*)alloc((size_t)(KVn * Dn) * HIDn * 2);
  bf16* wvT = (bf16*)alloc((size_t)(KVn * Dn) * HIDn * 2);
  bf16* woT = (bf16*)alloc((size_t)HIDn * HIDn * 2);
  bf16* Qbf = (bf16*)alloc((size_t)Bn * Hn  * Sn * Dn * 2);
  bf16* Kbf = (bf16*)alloc((size_t)Bn * KVn * Sn * Dn * 2);
  bf16* Vt  = (bf16*)alloc((size_t)Bn * KVn * Sn * Dn * 2);
  bf16* Ao  = (bf16*)alloc(NX * 2);

  // Stage 0: conversions / transposes
  cvt_bf16_kernel<<<(int)((NX + 255) / 256), 256, 0, stream>>>(x, xbf, (int)NX);
  transpose_bf16_kernel<<<(HIDn * HIDn + 255) / 256, 256, 0, stream>>>(wq, wqT, HIDn, HIDn);
  transpose_bf16_kernel<<<(HIDn * KVn * Dn + 255) / 256, 256, 0, stream>>>(wk, wkT, HIDn, KVn * Dn);
  transpose_bf16_kernel<<<(HIDn * KVn * Dn + 255) / 256, 256, 0, stream>>>(wv, wvT, HIDn, KVn * Dn);
  transpose_bf16_kernel<<<(HIDn * HIDn + 255) / 256, 256, 0, stream>>>(wo, woT, HIDn, HIDn);

  // Stage 1: QKV projection + RoPE (256 M-tiles x 160 N-tiles, 8 waves/block)
  qkv_rope_kernel<<<(Bn * Sn / 16) * 160 / 8, 256, 0, stream>>>(
      xbf, wqT, wkT, wvT, cosT, sinT, Qbf, Kbf, Vt);

  // Stage 2: flash attention (8192 waves, 4 waves/block)
  flash_attn_kernel<<<(Bn * Hn * (Sn / 16)) / 4, 128, 0, stream>>>(Qbf, Kbf, Vt, Ao);

  // Stage 3: output projection (256 x 128 tiles, 8 waves/block)
  outproj_kernel<<<((Bn * Sn / 16) * (HIDn / 16)) / 8, 256, 0, stream>>>(Ao, woT, out);
}
